// QuantizationLayer_80272938762716
// MI455X (gfx1250) — matrix operand order repaired and checked by reading, not compiled, not run on hardware
//
#include <hip/hip_runtime.h>
#include <hip/hip_bf16.h>
#include <math.h>

// ---------------- problem constants ----------------
#define B_      16
#define NEV     960000
#define S_      48
#define SEGLEN  20000
#define D_      256          // H == W == 256
#define S_IDX   3
#define E_IDX   38           // S - END_BIAS
#define N_ITER  34           // E_IDX - (S_IDX+1)

typedef _Float16 v16h __attribute__((ext_vector_type(16)));
typedef float    v8f  __attribute__((ext_vector_type(8)));

// =====================================================================
// Kernel 1: decode events -> packed u16 coords + per-(b,s) histograms.
// One block per (segment, batch): LDS-private histograms, zero global
// atomics. Streams the 122.9MB event buffer exactly once.
// =====================================================================
__global__ void __launch_bounds__(256) k_hist(const float* __restrict__ events,
                                              unsigned short* __restrict__ xy,
                                              float* __restrict__ alongX,
                                              float* __restrict__ alongY) {
  __shared__ unsigned hx[D_], hy[D_];
  const int s = blockIdx.x, b = blockIdx.y, tid = threadIdx.x;
  hx[tid] = 0u; hy[tid] = 0u;
  __syncthreads();
  const size_t base = (size_t)b * NEV + (size_t)s * SEGLEN;
  const float2* ev = reinterpret_cast<const float2*>(events) + base;
  unsigned short* xyo = xy + base;
  for (int i = tid; i < SEGLEN; i += 256) {
    if (i + 1024 < SEGLEN) __builtin_prefetch(&ev[i + 1024], 0, 0);
    float2 e = ev[i];
    int x = min(max((int)e.x, 0), 255);   // astype(int32): trunc, inputs >= 0
    int y = min(max((int)e.y, 0), 255);
    xyo[i] = (unsigned short)(x | (y << 8));
    atomicAdd(&hx[x], 1u);
    atomicAdd(&hy[y], 1u);
  }
  __syncthreads();
  const size_t o = ((size_t)b * S_ + s) * D_ + tid;
  alongX[o] = (float)hx[tid];
  alongY[o] = (float)hy[tid];
}

// =====================================================================
// Kernel 2: per batch (one block): cv = mean+3*std(ddof=1), clip, 5x5
// blur (SAME, zero pad), then m[s] = sum_d blur[s,d]*d / 20000 done as a
// WMMA matvec (A = 16x32 f16 blur tile, B[k,n] = weight(k) broadcast over
// n), aligned[s] = round(m[s]-128).
// Dynamic LDS: raw[12288] f32 + blur[12288] f32 = 96 KB.
// =====================================================================
__global__ void __launch_bounds__(256) k_axis(const float* __restrict__ along,
                                              float* __restrict__ meanOut,
                                              int* __restrict__ alignedOut) {
  extern __shared__ float sm[];
  float* raw = sm;              // 48*256
  float* blr = sm + S_ * D_;    // 48*256
  __shared__ double s_red[256], s_red2[256];
  __shared__ float  s_m[S_];
  const int b = blockIdx.x, tid = threadIdx.x;
  const float* A = along + (size_t)b * (S_ * D_);

  double ps = 0.0, pq = 0.0;
  for (int i = tid; i < S_ * D_; i += 256) {
    float v = A[i];
    raw[i] = v;
    ps += (double)v; pq += (double)v * (double)v;
  }
  s_red[tid] = ps; s_red2[tid] = pq;
  __syncthreads();
  for (int off = 128; off > 0; off >>= 1) {
    if (tid < off) { s_red[tid] += s_red[tid + off]; s_red2[tid] += s_red2[tid + off]; }
    __syncthreads();
  }
  const double sum = s_red[0], sq = s_red2[0];
  const double n  = (double)(S_ * D_);
  const double mean = sum / n;
  const double var  = (sq - sum * sum / n) / (n - 1.0);
  const float  cv   = (float)(mean + 3.0 * sqrt(var));

  // clip + 5x5 uniform blur (0.04), zero padding
  for (int i = tid; i < S_ * D_; i += 256) {
    const int s = i >> 8, d = i & 255;
    float acc = 0.0f;
    for (int di = -2; di <= 2; ++di) {
      const int ss = s + di;
      if (ss < 0 || ss >= S_) continue;
      for (int dj = -2; dj <= 2; ++dj) {
        const int dd = d + dj;
        if (dd < 0 || dd >= D_) continue;
        acc += fminf(raw[(ss << 8) + dd], cv);
      }
    }
    blr[i] = 0.04f * acc;
  }
  __syncthreads();

  // ---- WMMA matvec: wave 0 only (full EXEC on wave32) ----
  if (tid < 32) {
    const int lane = tid;
    const int lhi  = (lane >= 16) ? 1 : 0;
    for (int t = 0; t < 3; ++t) {                 // 3 tiles of 16 segments
      v8f c = {0.f, 0.f, 0.f, 0.f, 0.f, 0.f, 0.f, 0.f};
      for (int kk = 0; kk < 8; ++kk) {            // 8 K-slices of 32
        v16h av = {}, bv = {};
        const int row = 16 * t + (lane & 15);
#pragma unroll
        for (int j = 0; j < 16; ++j) {
          // 16-bit A 16x32 layout: VGPR v<4 -> K=2v(+1); v>=4 -> K=2v+8(+1);
          // lanes 16..31 add 8. Half 0/1 = even/odd K of the pair.
          const int v = j >> 1, hp = j & 1;
          const int K = ((v < 4) ? (2 * v) : (2 * v + 8)) + lhi * 8 + hp;
          av[j] = (_Float16)blr[(row << 8) + (kk << 5) + K];
          // B 32x16: same K order as A (j<8 -> K=j, j>=8 -> K=j+8; lanes 16..31 add 8).
          // B[k,n] = weight(k) = global d index, broadcast over n.
          const int Kb = ((j < 8) ? j : (j + 8)) + lhi * 8;
          bv[j] = (_Float16)(float)((kk << 5) + Kb);
        }
        c = __builtin_amdgcn_wmma_f32_16x16x32_f16(
              /*neg_a=*/false, av, /*neg_b=*/false, bv,
              /*c_mod=*/(short)0, c, /*reuse_a=*/false, /*reuse_b=*/false);
      }
      // D layout: VGPR r -> row r (lanes 0..15) / row r+8 (lanes 16..31),
      // col = lane&15; every column identical -> read col 0.
      if ((lane & 15) == 0) {
        const int mb = 16 * t + lhi * 8;
#pragma unroll
        for (int r = 0; r < 8; ++r) s_m[mb + r] = c[r] * (1.0f / (float)SEGLEN);
      }
    }
  }
  __syncthreads();
  if (tid < S_) {
    const float m = s_m[tid];
    meanOut[b * S_ + tid] = m;
    // round(m - start - (D/2 - start)) == round(m - 128); round-half-even
    alignedOut[b * S_ + tid] = (int)rintf(m - 128.0f);
  }
}

// =====================================================================
// Kernel 3: MAD outlier flags over 34 sliding windows of meanX/meanY.
// =====================================================================
__device__ inline void sort10(float* w) {
  for (int i = 1; i < 10; ++i) {
    const float v = w[i];
    int j = i - 1;
    while (j >= 0 && w[j] > v) { w[j + 1] = w[j]; --j; }
    w[j + 1] = v;
  }
}

__device__ inline bool out_flag(const float* m, int si) {
  float w[10];
  for (int k = 0; k < 10; ++k) w[k] = m[si + k];
  const float first = w[0];
  sort10(w);
  const float med = 0.5f * (w[4] + w[5]);
  float df[10];
  for (int k = 0; k < 10; ++k) df[k] = fabsf(w[k] - med);
  const float d0 = fabsf(first - med);
  sort10(df);
  const float mad = 0.5f * (df[4] + df[5]);
  return (0.6745f * d0 / mad) > 2.0f;   // IEEE: mad==0 -> inf/nan, matches jnp
}

__global__ void k_outlier(const float* __restrict__ mX,
                          const float* __restrict__ mY,
                          int* __restrict__ isOut) {
  const int b = blockIdx.x, t = threadIdx.x;
  if (t >= N_ITER) return;
  const int si = S_IDX + 1 + t;
  const bool f = out_flag(mX + b * S_, si) || out_flag(mY + b * S_, si);
  isOut[b * N_ITER + t] = f ? 1 : 0;
}

// =====================================================================
// Kernel 4: sequential per-batch scan. Whole 256x256 container lives in
// LDS as 2x u16 packed per u32 (128 KB) + double-buffered 16K-bit
// occupancy bitmaps (4 KB) -- possible thanks to CDNA5's 320 KB LDS/WGP.
// Coordinates re-read from the L2-resident packed u16 buffer.
// =====================================================================
__global__ void __launch_bounds__(1024) k_scan(const unsigned short* __restrict__ xy,
                                               const int* __restrict__ alX,
                                               const int* __restrict__ alY,
                                               const int* __restrict__ isOutArr,
                                               float* __restrict__ out) {
  extern __shared__ unsigned sm_u[];
  unsigned* cont = sm_u;                 // 32768 words = 65536 u16 cells
  unsigned* verA = sm_u + 32768;         // 512 words = 16384 bits
  unsigned* verB = sm_u + 32768 + 512;
  __shared__ int s_pc;
  __shared__ unsigned s_sum, s_sq;
  const int b = blockIdx.x, tid = threadIdx.x;
  const int T = 1024;
  const unsigned short* exy = xy + (size_t)b * NEV;

  for (int i = tid; i < 32768; i += T) cont[i] = 0u;
  for (int i = tid; i < 512; i += T) verA[i] = 0u;
  if (tid == 0) s_pc = 0;
  __syncthreads();

  // segment S_IDX = 3: container0 + ver0
  {
    const int ax = alX[b * S_ + S_IDX], ay = alY[b * S_ + S_IDX];
    for (int i = tid; i < SEGLEN; i += T) {
      const unsigned p = exy[S_IDX * SEGLEN + i];
      const int xs = min(max((int)(p & 255u) - ax, 0), 255);
      const int ys = min(max((int)(p >> 8) - ay, 0), 255);
      const int ic = xs + (ys << 8);
      const int iv = (xs >> 1) + ((ys >> 1) << 7);
      atomicAdd(&cont[ic >> 1], 1u << ((ic & 1) * 16));   // counts << 65536
      atomicOr(&verA[iv >> 5], 1u << (iv & 31));
    }
  }
  __syncthreads();
  { int pc = 0; for (int i = tid; i < 512; i += T) pc += __popc(verA[i]); atomicAdd(&s_pc, pc); }
  __syncthreads();
  int cntOld = s_pc;
  int cur = 0;
  bool stopped = false;

  for (int s = S_IDX + 1; s < E_IDX; ++s) {
    __syncthreads();                         // all read s_pc from prev iter
    if (tid == 0) s_pc = 0;
    unsigned* vc = cur ? verB : verA;
    unsigned* vn = cur ? verA : verB;
    for (int i = tid; i < 512; i += T) vn[i] = vc[i];
    __syncthreads();
    const int ax = alX[b * S_ + s], ay = alY[b * S_ + s];
    for (int i = tid; i < SEGLEN; i += T) {
      const unsigned p = exy[s * SEGLEN + i];
      const int xs = min(max((int)(p & 255u) - ax, 0), 255);
      const int ys = min(max((int)(p >> 8) - ay, 0), 255);
      const int iv = (xs >> 1) + ((ys >> 1) << 7);
      atomicOr(&vn[iv >> 5], 1u << (iv & 31));
    }
    __syncthreads();
    { int pc = 0; for (int i = tid; i < 512; i += T) pc += __popc(vn[i]); atomicAdd(&s_pc, pc); }
    __syncthreads();
    const int cntNew = s_pc;                                 // uniform
    const bool isOut = isOutArr[b * N_ITER + (s - S_IDX - 1)] != 0;
    const bool brk = ((float)(cntNew - cntOld) / (float)cntNew) < 0.1f;
    if (!isOut && brk) stopped = true;                       // !stopped here
    if (stopped) break;                                      // frozen forever
    if (!isOut) {                                            // do = commit
      for (int i = tid; i < SEGLEN; i += T) {
        const unsigned p = exy[s * SEGLEN + i];
        const int xs = min(max((int)(p & 255u) - ax, 0), 255);
        const int ys = min(max((int)(p >> 8) - ay, 0), 255);
        const int ic = xs + (ys << 8);
        atomicAdd(&cont[ic >> 1], 1u << ((ic & 1) * 16));
      }
      cur ^= 1;
      cntOld = cntNew;
    }
  }
  __syncthreads();

  // cv over container (exact integer sums), then clip/scale out
  if (tid == 0) { s_sum = 0u; s_sq = 0u; }
  __syncthreads();
  { unsigned ls = 0u, lq = 0u;
    for (int i = tid; i < 32768; i += T) {
      const unsigned w = cont[i];
      const unsigned c0 = w & 0xFFFFu, c1 = w >> 16;
      ls += c0 + c1; lq += c0 * c0 + c1 * c1;
    }
    atomicAdd(&s_sum, ls); atomicAdd(&s_sq, lq); }
  __syncthreads();
  const double sum = (double)s_sum, sq = (double)s_sq;
  const double mean = sum / 65536.0;
  const double var  = (sq - sum * sum / 65536.0) / 65535.0;
  const float  cvf  = (float)(mean + 3.0 * sqrt(var));
  const float  inv  = 1.0f / cvf;
  float* ob = out + (size_t)b * 65536;
  for (int i = tid; i < 65536; i += T) {
    const unsigned w = cont[i >> 1];
    const float c = (float)((w >> ((i & 1) * 16)) & 0xFFFFu);
    ob[i] = fminf(c, cvf) * inv;
  }
}

// =====================================================================
extern "C" void kernel_launch(void* const* d_in, const int* in_sizes, int n_in,
                              void* d_out, int out_size, void* d_ws, size_t ws_size,
                              hipStream_t stream) {
  (void)in_sizes; (void)n_in; (void)out_size; (void)ws_size;
  const float* events = (const float*)d_in[0];
  float* out = (float*)d_out;
  char* ws = (char*)d_ws;

  size_t off = 0;
  auto take = [&](size_t bytes) -> void* {
    void* p = ws + off;
    off += (bytes + 255) & ~(size_t)255;
    return p;
  };
  unsigned short* xy = (unsigned short*)take((size_t)B_ * NEV * sizeof(unsigned short)); // 30.72 MB
  float* alongX   = (float*)take((size_t)B_ * S_ * D_ * sizeof(float));
  float* alongY   = (float*)take((size_t)B_ * S_ * D_ * sizeof(float));
  float* meanX    = (float*)take((size_t)B_ * S_ * sizeof(float));
  float* meanY    = (float*)take((size_t)B_ * S_ * sizeof(float));
  int*   alignedX = (int*)take((size_t)B_ * S_ * sizeof(int));
  int*   alignedY = (int*)take((size_t)B_ * S_ * sizeof(int));
  int*   isOut    = (int*)take((size_t)B_ * N_ITER * sizeof(int));

  k_hist<<<dim3(S_, B_), 256, 0, stream>>>(events, xy, alongX, alongY);

  const size_t axisLds = 2u * S_ * D_ * sizeof(float);       // 96 KB
  k_axis<<<B_, 256, axisLds, stream>>>(alongX, meanX, alignedX);
  k_axis<<<B_, 256, axisLds, stream>>>(alongY, meanY, alignedY);

  k_outlier<<<B_, 64, 0, stream>>>(meanX, meanY, isOut);

  const size_t scanLds = (32768u + 1024u) * sizeof(unsigned); // 132 KB
  k_scan<<<B_, 1024, scanLds, stream>>>(xy, alignedX, alignedY, isOut, out);
}
